// QuantLinear_29875792511250
// MI455X (gfx1250) — compile-verified, matrix operand order checked by reading, not run on hardware
//
#include <hip/hip_runtime.h>
#include <hip/hip_bf16.h>

typedef __attribute__((ext_vector_type(16))) __bf16 v16bf;
typedef __attribute__((ext_vector_type(8)))  float  v8f;
typedef __attribute__((ext_vector_type(8)))  __bf16 bf16x8;
typedef __attribute__((ext_vector_type(4)))  __bf16 bf16x4;

// Problem constants (match the reference)
#define KDIM 4096
#define NDIM 4096
#define MDIM 8192        // 4 * 2048
#define GROUP 128

// Tiling
#define BM 128
#define BN 128
#define BK 64
#define LDK 72           // padded LDS K-stride (halves) to spread banks
#define TILE_HALVES (BM * LDK)

__global__ __launch_bounds__(256)
void qlinear_wmma_bf16(const float* __restrict__ x,
                       const int*   __restrict__ qweight,   // [K/8, N] packed nibbles along K
                       const float* __restrict__ scales,    // [K/GROUP, N]
                       const int*   __restrict__ qzeros,    // [K/GROUP, N/8] packed nibbles along N
                       const float* __restrict__ bias,      // [N]
                       float*       __restrict__ out)       // [M, N]
{
    // Double-buffered tiles: [2][BM][LDK] halves each
    __shared__ __bf16 As[2 * TILE_HALVES];   // A: row-major per M row [m][k]
    __shared__ __bf16 Bs[2 * TILE_HALVES];   // B: K-major per N column [n][k]

    const int tid  = threadIdx.x;
    const int lane = tid & 31;
    const int wave = tid >> 5;
    const int wm   = (wave & 3) * 32;   // wave M offset within block tile
    const int wn   = (wave >> 2) * 64;  // wave N offset within block tile
    const int bn0  = blockIdx.x * BN;
    const int bm0  = blockIdx.y * BM;

    // Staging thread mappings (fixed across slabs)
    const int a_c  = (tid & 15) * 4;    // 16 threads * float4 = 64 floats per A row
    const int a_r0 = tid >> 4;          // 16 rows per pass, 8 passes
    // B: 1024 packed words per slab, 4 per thread
    int b_n[4], b_kp[4];
    #pragma unroll
    for (int it = 0; it < 4; ++it) {
        const int idx = tid + it * 256;
        b_n[it]  = idx & 127;           // column within tile (coalesced over tid)
        b_kp[it] = idx >> 7;            // 0..7 packed K-rows (8 nibbles each)
    }

    v8f acc[2][4] = {};

    const int lm = lane & 15;           // fragment row/col selector
    const int kl = (lane >> 4) << 4;    // K half: lanes 0-15 -> K 0..15, lanes 16-31 -> K 16..31

    // ---------------- helpers as lambdas ----------------
    auto load_slab = [&](int k0, float4 xa[8], int wq[4], int zw[4], float sc[4]) {
        #pragma unroll
        for (int rr = 0; rr < 8; ++rr) {
            const int row = rr * 16 + a_r0;
            xa[rr] = *(const float4*)(x + (size_t)(bm0 + row) * KDIM + k0 + a_c);
        }
        const int g = k0 >> 7;          // GROUP=128, slab fits in one group
        #pragma unroll
        for (int it = 0; it < 4; ++it) {
            const int ng = bn0 + b_n[it];
            wq[it] = qweight[(size_t)((k0 >> 3) + b_kp[it]) * NDIM + ng];
            zw[it] = qzeros[(size_t)g * (NDIM / 8) + (ng >> 3)];
            sc[it] = scales[(size_t)g * NDIM + ng];
        }
    };

    auto store_slab = [&](int buf, const float4 xa[8], const int wq[4],
                          const int zw[4], const float sc[4]) {
        __bf16* Asb = As + buf * TILE_HALVES;
        __bf16* Bsb = Bs + buf * TILE_HALVES;
        #pragma unroll
        for (int rr = 0; rr < 8; ++rr) {
            const int row = rr * 16 + a_r0;
            bf16x4 h;
            h.x = (__bf16)xa[rr].x; h.y = (__bf16)xa[rr].y;
            h.z = (__bf16)xa[rr].z; h.w = (__bf16)xa[rr].w;
            *(bf16x4*)&Asb[row * LDK + a_c] = h;
        }
        #pragma unroll
        for (int it = 0; it < 4; ++it) {
            const int ng = bn0 + b_n[it];
            const float s  = sc[it];
            const float zs = (float)(((zw[it] >> ((ng & 7) * 4)) & 15) + 1) * s;
            bf16x8 h;
            #pragma unroll
            for (int i = 0; i < 8; ++i) {
                const float q = (float)((wq[it] >> (4 * i)) & 15);
                h[i] = (__bf16)(q * s - zs);   // (q - (z+1)) * s, folded to FMA
            }
            *(bf16x8*)&Bsb[b_n[it] * LDK + b_kp[it] * 8] = h;
        }
    };

    auto compute_slab = [&](int buf) {
        const __bf16* Asb = As + buf * TILE_HALVES;
        const __bf16* Bsb = Bs + buf * TILE_HALVES;
        #pragma unroll
        for (int kk = 0; kk < BK; kk += 32) {
            v16bf a[2], b[4];
            #pragma unroll
            for (int tm = 0; tm < 2; ++tm)
                a[tm] = *(const v16bf*)&Asb[(wm + tm * 16 + lm) * LDK + kk + kl];
            #pragma unroll
            for (int tn = 0; tn < 4; ++tn)
                b[tn] = *(const v16bf*)&Bsb[(wn + tn * 16 + lm) * LDK + kk + kl];
            #pragma unroll
            for (int tm = 0; tm < 2; ++tm) {
                #pragma unroll
                for (int tn = 0; tn < 4; ++tn) {
                    acc[tm][tn] = __builtin_amdgcn_wmma_f32_16x16x32_bf16(
                        false, a[tm], false, b[tn],
                        (short)0, acc[tm][tn], false, false);
                }
            }
        }
    };
    // ----------------------------------------------------

    // Prologue: stage slab 0 into buffer 0
    {
        float4 xa[8]; int wq[4]; int zw[4]; float sc[4];
        load_slab(0, xa, wq, zw, sc);
        store_slab(0, xa, wq, zw, sc);
    }
    __syncthreads();

    // Software-pipelined main loop: one barrier per slab.
    // Iter i: global-load slab i+1 -> compute slab i (WMMA hides load latency)
    //         -> dequant+LDS-store slab i+1 into other buffer -> barrier.
    for (int k0 = 0; k0 < KDIM; k0 += BK) {
        const int  cur      = (k0 >> 6) & 1;
        const bool has_next = (k0 + BK) < KDIM;

        float4 xa[8]; int wq[4]; int zw[4]; float sc[4];
        if (has_next) {
            load_slab(k0 + BK, xa, wq, zw, sc);
            if (k0 + 2 * BK < KDIM) {   // warm L2 two slabs ahead (gfx1250 prefetch path)
                __builtin_prefetch(qweight + (size_t)(((k0 + 2 * BK) >> 3) + (tid >> 7)) * NDIM
                                           + bn0 + (tid & 127), 0, 3);
            }
        }

        compute_slab(cur);

        if (has_next) {
            store_slab(cur ^ 1, xa, wq, zw, sc);
        }
        __syncthreads();
    }

    // ---- Epilogue: bias add + store (C/D layout: VGPR v -> M = v + 8*(lane>=16), lane%16 -> N) ----
    #pragma unroll
    for (int tn = 0; tn < 4; ++tn) {
        const int col = bn0 + wn + tn * 16 + lm;
        const float bv = bias[col];
        #pragma unroll
        for (int tm = 0; tm < 2; ++tm) {
            const int rowbase = bm0 + wm + tm * 16 + (lane >> 4) * 8;
            #pragma unroll
            for (int v = 0; v < 8; ++v) {
                out[(size_t)(rowbase + v) * NDIM + col] = acc[tm][tn][v] + bv;
            }
        }
    }
}

extern "C" void kernel_launch(void* const* d_in, const int* in_sizes, int n_in,
                              void* d_out, int out_size, void* d_ws, size_t ws_size,
                              hipStream_t stream) {
    const float* x       = (const float*)d_in[0];   // [4, 2048, 4096] fp32
    const int*   qweight = (const int*)d_in[1];     // [512, 4096] int32
    const float* scales  = (const float*)d_in[2];   // [32, 4096] fp32
    const int*   qzeros  = (const int*)d_in[3];     // [32, 512] int32
    const float* bias    = (const float*)d_in[4];   // [4096] fp32
    float*       out     = (float*)d_out;           // [8192, 4096] fp32

    dim3 grid(NDIM / BN, MDIM / BM);  // 32 x 64 blocks
    dim3 block(256);
    qlinear_wmma_bf16<<<grid, block, 0, stream>>>(x, qweight, scales, qzeros, bias, out);
}
